// RWKVBlock_v2_85066122264907
// MI455X (gfx1250) — matrix operand order repaired and checked by reading, not compiled, not run on hardware
//
#include <hip/hip_runtime.h>
#include <hip/hip_bf16.h>
#include <math.h>

// Problem constants (match reference)
#define NB   4        // batch
#define CC   64       // channels
#define C4N  256      // 4*CC
#define NHN  8        // heads
#define NN   32       // head dim
#define HD   64
#define WD   64
#define TT   4096     // HD*WD
#define EPSF 1e-5f

typedef _Float16 v16h __attribute__((ext_vector_type(16)));
typedef float    v8f  __attribute__((ext_vector_type(8)));

// ---------------------------------------------------------------------------
// WMMA fragment swizzles (CDNA5 ISA 7.12.2, 16-bit layouts, wave32)
// A-matrix 16x32 f16: lane L (M = L%16), VGPR p<4: K = 2p + (L/16)*8 + half
//                     VGPR p>=4:        K = 16 + 2(p-4) + (L/16)*8 + half
__device__ __forceinline__ size_t aswz(int m, int k) {
  int tmi = m >> 4, tk = k >> 5, kl = k & 31;
  int hi = kl >> 4, lhalf = (kl >> 3) & 1, q = (kl >> 1) & 3, rb = kl & 1;
  int L = (m & 15) + (lhalf << 4);
  int e = (((hi << 2) + q) << 1) + rb;          // element index in v16h
  return (((size_t)tmi * 8 + tk) * 32 + L) * 16 + e;
}
// B-matrix 32x16 f16: lane L (N = L%16), same K order as A:
// elem e<8: K = (L/16)*8 + e; e>=8: K = 16 + (L/16)*8 + (e-8)
__device__ __forceinline__ size_t bswz(int n, int k) {
  int tn = n >> 4, tk = k >> 5, kl = k & 31;
  int lhalf = (kl >> 3) & 1;
  int e = ((kl >> 4) << 3) | (kl & 7);
  int L = (n & 15) + (lhalf << 4);
  return (((size_t)tn * 8 + tk) * 32 + L) * 16 + e;
}

// ---------------------------------------------------------------------------
// 1) cond fusion (1x1 conv + depthwise 3x3, recomputing the 1x1 per tap) then
//    4-direction cross-scan into z (B, C4, T)
__global__ void k_fuse_scan(const float* __restrict__ x, const float* __restrict__ cond,
                            const float* __restrict__ w1, const float* __restrict__ b1,
                            const float* __restrict__ w2, float* __restrict__ z)
{
  int idx = blockIdx.x * blockDim.x + threadIdx.x;        // B*C*H*W
  int w = idx & 63, h = (idx >> 6) & 63, c = (idx >> 12) & 63, b = idx >> 18;
  float acc = 0.f;
  #pragma unroll
  for (int dy = -1; dy <= 1; ++dy) {
    #pragma unroll
    for (int dx = -1; dx <= 1; ++dx) {
      int hh = h + dy, ww = w + dx;
      if (hh >= 0 && hh < HD && ww >= 0 && ww < WD) {
        float cf = b1[c];
        #pragma unroll
        for (int i = 0; i < 3; ++i)
          cf += cond[((b * 3 + i) * HD + hh) * WD + ww] * w1[c * 3 + i];
        acc += cf * w2[c * 9 + (dy + 1) * 3 + (dx + 1)];
      }
    }
  }
  float xc = x[idx] + acc;
  int t0 = h * WD + w, t1 = w * HD + h;
  float* zb = z + (size_t)b * C4N * TT;
  zb[(size_t)(c)       * TT + t0]            = xc;
  zb[(size_t)(64 + c)  * TT + t1]            = xc;
  zb[(size_t)(128 + c) * TT + (TT - 1 - t0)] = xc;
  zb[(size_t)(192 + c) * TT + (TT - 1 - t1)] = xc;
}

// ---------------------------------------------------------------------------
// 2) GroupNorm(1, C4) over (C4, T) per sample: two-stage reduction
__global__ void k_gn_partial(const float* __restrict__ z, float* __restrict__ part)
{
  int b = blockIdx.x >> 6, chunk = blockIdx.x & 63;       // NB*64 blocks x 256 thr
  const float* zb = z + (size_t)b * C4N * TT + (size_t)chunk * 16384;
  float s = 0.f, sq = 0.f;
  for (int i = threadIdx.x; i < 16384; i += 256) { float v = zb[i]; s += v; sq += v * v; }
  __shared__ float sh[512];
  sh[threadIdx.x] = s; sh[256 + threadIdx.x] = sq;
  __syncthreads();
  for (int off = 128; off > 0; off >>= 1) {
    if (threadIdx.x < off) { sh[threadIdx.x] += sh[threadIdx.x + off];
                             sh[256 + threadIdx.x] += sh[256 + threadIdx.x + off]; }
    __syncthreads();
  }
  if (threadIdx.x == 0) { part[blockIdx.x * 2] = sh[0]; part[blockIdx.x * 2 + 1] = sh[256]; }
}

__global__ void k_gn_final(const float* __restrict__ part, float* __restrict__ stats)
{
  int b = blockIdx.x;                                     // NB blocks x 64 thr
  float s = part[(b * 64 + threadIdx.x) * 2];
  float sq = part[(b * 64 + threadIdx.x) * 2 + 1];
  __shared__ float sh[128];
  sh[threadIdx.x] = s; sh[64 + threadIdx.x] = sq;
  __syncthreads();
  for (int off = 32; off > 0; off >>= 1) {
    if (threadIdx.x < off) { sh[threadIdx.x] += sh[threadIdx.x + off];
                             sh[64 + threadIdx.x] += sh[64 + threadIdx.x + off]; }
    __syncthreads();
  }
  if (threadIdx.x == 0) {
    float n = (float)((size_t)C4N * TT);
    float mean = sh[0] / n, var = sh[64] / n - (sh[0] / n) * (sh[0] / n);
    stats[b * 2] = mean; stats[b * 2 + 1] = rsqrtf(var + EPSF);
  }
}

// ---------------------------------------------------------------------------
// 3) normalize + token-shift mix, emit 4 f16 A-fragments (mu_r/k/v/g)
__global__ void k_mix4(const float* __restrict__ z, const float* __restrict__ stats,
                       const float* __restrict__ g, const float* __restrict__ bb,
                       const float* __restrict__ mur, const float* __restrict__ muk,
                       const float* __restrict__ muv, const float* __restrict__ mug,
                       _Float16* __restrict__ A0, _Float16* __restrict__ A1,
                       _Float16* __restrict__ A2, _Float16* __restrict__ A3)
{
  int idx = blockIdx.x * 256 + threadIdx.x;               // B*T*C4
  int c = idx & 255, t = (idx >> 8) & 4095, b = idx >> 20;
  float m = stats[b * 2], inv = stats[b * 2 + 1];
  float gc = g[c], bc = bb[c];
  const float* zc = z + ((size_t)b * C4N + c) * (size_t)TT;
  float xn = (zc[t] - m) * inv * gc + bc;
  float xp = (t > 0) ? ((zc[t - 1] - m) * inv * gc + bc) : 0.f;
  int row = b * TT + t;
  size_t si = aswz(row, c);
  A0[si] = (_Float16)(xn + (xp - xn) * mur[c]);
  A1[si] = (_Float16)(xn + (xp - xn) * muk[c]);
  A2[si] = (_Float16)(xn + (xp - xn) * muv[c]);
  A3[si] = (_Float16)(xn + (xp - xn) * mug[c]);
}

// cmix variant: 2 outputs (fmu_k, fmu_r)
__global__ void k_mix2(const float* __restrict__ z, const float* __restrict__ stats,
                       const float* __restrict__ g, const float* __restrict__ bb,
                       const float* __restrict__ muk, const float* __restrict__ mur,
                       _Float16* __restrict__ A0, _Float16* __restrict__ A1)
{
  int idx = blockIdx.x * 256 + threadIdx.x;
  int c = idx & 255, t = (idx >> 8) & 4095, b = idx >> 20;
  float m = stats[b * 2], inv = stats[b * 2 + 1];
  float gc = g[c], bc = bb[c];
  const float* zc = z + ((size_t)b * C4N + c) * (size_t)TT;
  float xn = (zc[t] - m) * inv * gc + bc;
  float xp = (t > 0) ? ((zc[t - 1] - m) * inv * gc + bc) : 0.f;
  int row = b * TT + t;
  size_t si = aswz(row, c);
  A0[si] = (_Float16)(xn + (xp - xn) * muk[c]);
  A1[si] = (_Float16)(xn + (xp - xn) * mur[c]);
}

// weight (256x256 f32, row = input k, col = output n) -> f16 B-fragment
__global__ void k_wconv(const float* __restrict__ W, _Float16* __restrict__ Wsw)
{
  int idx = blockIdx.x * 256 + threadIdx.x;               // 65536
  int n = idx & 255, k = idx >> 8;
  Wsw[bswz(n, k)] = (_Float16)W[k * 256 + n];
}

// ---------------------------------------------------------------------------
// 4) WMMA GEMM: (16384 x 256) = Asw(16384x256,f16) @ Bsw(256x256,f16), f32 acc.
//    One wave computes a 16x64 tile (A frag reused across 4 B frags).
//    op: 0 none, 1 silu, 2 sigmoid, 3 relu^2
//    mode: 0 store row-major (M,256) f32
//          1 accumulate transposed into z[b,c,t]
//          2 store f16 directly into an A-fragment buffer (feeds next GEMM)
__global__ void k_gemm(const _Float16* __restrict__ Asw, const _Float16* __restrict__ Bsw,
                       float* __restrict__ Out, int op, int mode)
{
  const int lane = threadIdx.x;                           // 0..31, full wave
  const int tm  = blockIdx.x;                             // 16-row tile
  const int tn0 = blockIdx.y * 4;                         // 4 x 16-col tiles
  v8f acc[4] = {v8f{}, v8f{}, v8f{}, v8f{}};
  for (int kb = 0; kb < 8; ++kb) {                        // K = 256 = 8 x 32
    v16h a = *(const v16h*)(Asw + (((size_t)tm * 8 + kb) * 32 + lane) * 16);
    #pragma unroll
    for (int j = 0; j < 4; ++j) {
      v16h bf = *(const v16h*)(Bsw + (((size_t)(tn0 + j) * 8 + kb) * 32 + lane) * 16);
      acc[j] = __builtin_amdgcn_wmma_f32_16x16x32_f16(
          false, a, false, bf, (short)0, acc[j], false, false);
    }
  }
  const int rl = lane >> 4, cl = lane & 15;
  #pragma unroll
  for (int j = 0; j < 4; ++j) {
    #pragma unroll
    for (int r = 0; r < 8; ++r) {
      int m = tm * 16 + r + rl * 8;                       // D layout: VGPR r -> M=r(+8)
      int n = (tn0 + j) * 16 + cl;
      float val = acc[j][r];
      if      (op == 1) val = val / (1.f + __expf(-val));        // silu
      else if (op == 2) val = 1.f / (1.f + __expf(-val));        // sigmoid
      else if (op == 3) { val = fmaxf(val, 0.f); val = val * val; } // relu^2
      if (mode == 0) Out[(size_t)m * 256 + n] = val;
      else if (mode == 1) {
        int b = m >> 12, t = m & 4095;
        Out[((size_t)b * 256 + n) * TT + t] += val;
      } else {
        ((_Float16*)Out)[aswz(m, n)] = (_Float16)val;
      }
    }
  }
}

// ---------------------------------------------------------------------------
// 5) RWKV-v5 recurrence: one wave32 per (b, head). Lane j holds column j of the
//    32x32 state S (key i x value j) in 32 VGPRs.
//    y_j = sum_i r_i S_ij + (sum_i r_i u_i k_i) * v_j   (u-term factorized)
//    S_ij <- S_ij * w_i + k_i * v_j
//    Next step's r/k/v loads are issued before the FMA chain (software pipeline).
__global__ void k_scan(const float* __restrict__ r, const float* __restrict__ k,
                       const float* __restrict__ v, const float* __restrict__ decay,
                       const float* __restrict__ u, float* __restrict__ y)
{
  int b = blockIdx.x >> 3, h = blockIdx.x & 7;
  int j = threadIdx.x;
  float wj = __expf(-__expf(decay[h * 32 + j]));          // per key channel
  float uj = u[h * 32 + j];
  float S[32];
  #pragma unroll
  for (int i = 0; i < 32; ++i) S[i] = 0.f;
  size_t base = (size_t)b * TT * 256 + h * 32 + j;
  float rv = r[base], kv = k[base], vv = v[base];
  for (int t = 0; t < TT; ++t) {
    // prefetch next step (clamped; wave-uniform index)
    int tn = (t + 1 < TT) ? (t + 1) : (TT - 1);
    size_t on = base + (size_t)tn * 256;
    float rn = r[on], kn = k[on], vn = v[on];
    // scalar bonus term: dot = sum_i r_i * u_i * k_i (all lanes get the sum)
    float ruk = rv * kv * uj;
    #pragma unroll
    for (int off = 16; off > 0; off >>= 1) ruk += __shfl_xor(ruk, off, 32);
    float acc = ruk * vv;
    #pragma unroll
    for (int i = 0; i < 32; ++i) {
      float ri = __shfl(rv, i, 32);
      float ki = __shfl(kv, i, 32);
      float wi = __shfl(wj, i, 32);
      acc += ri * S[i];                                   // r . S (pre-update)
      S[i] = S[i] * wi + ki * vv;                         // decay + outer-product
    }
    y[base + (size_t)t * 256] = acc;
    rv = rn; kv = kn; vv = vn;
  }
}

// ---------------------------------------------------------------------------
// 6) per-head GroupNorm (NH groups of 32) + gate, emit f16 A-fragment for Wo
__global__ void k_headnorm(const float* __restrict__ y, const float* __restrict__ g,
                           const float* __restrict__ lg, const float* __restrict__ lb,
                           _Float16* __restrict__ A0)
{
  int bi = blockIdx.x;                                    // b*T + t
  int c = threadIdx.x;                                    // 256 thr = 8 waves = 8 heads
  float val = y[(size_t)bi * 256 + c];
  float s = val, sq = val * val;
  #pragma unroll
  for (int off = 16; off > 0; off >>= 1) {
    s  += __shfl_xor(s, off, 32);
    sq += __shfl_xor(sq, off, 32);
  }
  float mean = s * (1.f / 32.f);
  float var  = sq * (1.f / 32.f) - mean * mean;
  float nv = (val - mean) * rsqrtf(var + EPSF) * lg[c] + lb[c];
  float gated = nv * g[(size_t)bi * 256 + c];
  A0[aswz(bi, c)] = (_Float16)gated;
}

// ---------------------------------------------------------------------------
// 7) cmix combine: z[b,c,t] += sigmoid_r * (kk @ fWv)
__global__ void k_combine(const float* __restrict__ sr, const float* __restrict__ kkv,
                          float* __restrict__ z)
{
  int idx = blockIdx.x * 256 + threadIdx.x;
  int c = idx & 255, t = (idx >> 8) & 4095, b = idx >> 20;
  z[((size_t)b * 256 + c) * TT + t] += sr[idx] * kkv[idx];
}

// ---------------------------------------------------------------------------
// 8) cross-merge + channel LayerNorm + residual
__global__ void k_merge(const float* __restrict__ z, const float* __restrict__ x,
                        const float* __restrict__ g, const float* __restrict__ bb,
                        float* __restrict__ out)
{
  int pix = blockIdx.x;                                   // B*H*W blocks x 64 thr
  int b = pix >> 12, p = pix & 4095;
  int h = p >> 6, w = p & 63;
  int c = threadIdx.x;
  int t0 = h * WD + w, t1 = w * HD + h;
  const float* zb = z + (size_t)b * C4N * TT;
  float ym = zb[(size_t)c * TT + t0]
           + zb[(size_t)(128 + c) * TT + (TT - 1 - t0)]
           + zb[(size_t)(64 + c)  * TT + t1]
           + zb[(size_t)(192 + c) * TT + (TT - 1 - t1)];
  __shared__ float sh[128];
  sh[c] = ym; sh[64 + c] = ym * ym;
  __syncthreads();
  for (int off = 32; off > 0; off >>= 1) {
    if (c < off) { sh[c] += sh[c + off]; sh[64 + c] += sh[64 + c + off]; }
    __syncthreads();
  }
  float mean = sh[0] * (1.f / 64.f);
  float var  = sh[64] * (1.f / 64.f) - mean * mean;
  float o = (ym - mean) * rsqrtf(var + EPSF) * g[c] + bb[c]
          + x[((size_t)(b * 64 + c) * 64 + h) * 64 + w];
  out[(size_t)(b * 64 + c) * TT + t0] = o;
}

// ---------------------------------------------------------------------------
extern "C" void kernel_launch(void* const* d_in, const int* in_sizes, int n_in,
                              void* d_out, int out_size, void* d_ws, size_t ws_size,
                              hipStream_t stream)
{
  const float* x     = (const float*)d_in[0];
  const float* cond  = (const float*)d_in[1];
  const float* fw1   = (const float*)d_in[2];
  const float* fb1   = (const float*)d_in[3];
  const float* fw2   = (const float*)d_in[4];
  const float* ln1g  = (const float*)d_in[5];
  const float* ln1b  = (const float*)d_in[6];
  const float* ln2g  = (const float*)d_in[7];
  const float* ln2b  = (const float*)d_in[8];
  const float* mur   = (const float*)d_in[9];
  const float* muk   = (const float*)d_in[10];
  const float* muv   = (const float*)d_in[11];
  const float* mug   = (const float*)d_in[12];
  const float* decay = (const float*)d_in[13];
  const float* faaaa = (const float*)d_in[14];
  const float* Wr    = (const float*)d_in[15];
  const float* Wk    = (const float*)d_in[16];
  const float* Wv    = (const float*)d_in[17];
  const float* Wg    = (const float*)d_in[18];
  const float* Wo    = (const float*)d_in[19];
  const float* lnxg  = (const float*)d_in[20];
  const float* lnxb  = (const float*)d_in[21];
  const float* fmuk  = (const float*)d_in[22];
  const float* fmur  = (const float*)d_in[23];
  const float* fWk   = (const float*)d_in[24];
  const float* fWv   = (const float*)d_in[25];
  const float* fWr   = (const float*)d_in[26];
  const float* outg  = (const float*)d_in[27];
  const float* outb  = (const float*)d_in[28];

  char* ws = (char*)d_ws;
  const size_t MB = 1024 * 1024;
  float*    z    = (float*)(ws);                // 16 MB  (B,C4,T)
  float*    rbuf = (float*)(ws + 16 * MB);      // 16 MB  (B,T,C4)
  float*    kbuf = (float*)(ws + 32 * MB);
  float*    vbuf = (float*)(ws + 48 * MB);
  float*    gbuf = (float*)(ws + 64 * MB);
  float*    ybuf = (float*)(ws + 80 * MB);
  _Float16* A0   = (_Float16*)(ws + 96 * MB);   // 8 MB f16 A fragments
  _Float16* A1   = (_Float16*)(ws + 104 * MB);
  _Float16* A2   = (_Float16*)(ws + 112 * MB);
  _Float16* A3   = (_Float16*)(ws + 120 * MB);
  _Float16* W16  = (_Float16*)(ws + 128 * MB);  // 8 x 65536 halves = 1 MB
  float*    part  = (float*)(ws + 130 * MB);
  float*    stats = (float*)(ws + 130 * MB + 4096);

  const int NEL = NB * TT * C4N;                // 4,194,304

  // weights -> f16 B-fragments (order: Wr,Wk,Wv,Wg,Wo,fWk,fWv,fWr)
  k_wconv<<<256, 256, 0, stream>>>(Wr,  W16 + 0 * 65536);
  k_wconv<<<256, 256, 0, stream>>>(Wk,  W16 + 1 * 65536);
  k_wconv<<<256, 256, 0, stream>>>(Wv,  W16 + 2 * 65536);
  k_wconv<<<256, 256, 0, stream>>>(Wg,  W16 + 3 * 65536);
  k_wconv<<<256, 256, 0, stream>>>(Wo,  W16 + 4 * 65536);
  k_wconv<<<256, 256, 0, stream>>>(fWk, W16 + 5 * 65536);
  k_wconv<<<256, 256, 0, stream>>>(fWv, W16 + 6 * 65536);
  k_wconv<<<256, 256, 0, stream>>>(fWr, W16 + 7 * 65536);

  // fuse conv + cross-scan
  k_fuse_scan<<<(NB * CC * HD * WD) / 256, 256, 0, stream>>>(x, cond, fw1, fb1, fw2, z);

  // GN1 + token-shift mixes
  k_gn_partial<<<NB * 64, 256, 0, stream>>>(z, part);
  k_gn_final<<<NB, 64, 0, stream>>>(part, stats);
  k_mix4<<<NEL / 256, 256, 0, stream>>>(z, stats, ln1g, ln1b, mur, muk, muv, mug,
                                        A0, A1, A2, A3);

  // r,k,v,g projections (WMMA)
  dim3 gg(TT * NB / 16, 4);                      // 1024 x 4 tiles, 1 wave each
  k_gemm<<<gg, 32, 0, stream>>>(A0, W16 + 0 * 65536, rbuf, 0, 0);
  k_gemm<<<gg, 32, 0, stream>>>(A1, W16 + 1 * 65536, kbuf, 0, 0);
  k_gemm<<<gg, 32, 0, stream>>>(A2, W16 + 2 * 65536, vbuf, 0, 0);
  k_gemm<<<gg, 32, 0, stream>>>(A3, W16 + 3 * 65536, gbuf, 1, 0);   // silu gate

  // linear-attention recurrence (32 independent wave32 streams)
  k_scan<<<NB * NHN, 32, 0, stream>>>(rbuf, kbuf, vbuf, decay, faaaa, ybuf);

  // head GroupNorm + gate, then Wo projection accumulated transposed into z
  k_headnorm<<<NB * TT, 256, 0, stream>>>(ybuf, gbuf, lnxg, lnxb, A0);
  k_gemm<<<gg, 32, 0, stream>>>(A0, W16 + 4 * 65536, z, 0, 1);

  // channel-mix: GN2, mixes, kk = relu(.)^2 -> f16 frag, kkv, sigmoid(r), combine
  k_gn_partial<<<NB * 64, 256, 0, stream>>>(z, part);
  k_gn_final<<<NB, 64, 0, stream>>>(part, stats);
  k_mix2<<<NEL / 256, 256, 0, stream>>>(z, stats, ln2g, ln2b, fmuk, fmur, A0, A1);
  k_gemm<<<gg, 32, 0, stream>>>(A0, W16 + 5 * 65536, (float*)A2, 3, 2); // relu^2 -> frag
  k_gemm<<<gg, 32, 0, stream>>>(A2, W16 + 6 * 65536, kbuf, 0, 0);       // kk @ fWv
  k_gemm<<<gg, 32, 0, stream>>>(A1, W16 + 7 * 65536, vbuf, 2, 0);       // sigmoid
  k_combine<<<NEL / 256, 256, 0, stream>>>(vbuf, kbuf, z);

  // cross-merge + out LayerNorm + residual
  k_merge<<<NB * HD * WD, 64, 0, stream>>>(z, x, outg, outb, (float*)d_out);
}
